// LTCAttentionCell_28389733827178
// MI455X (gfx1250) — compile-verified, matrix operand order checked by reading, not run on hardware
//
#include <hip/hip_runtime.h>
#include <hip/hip_bf16.h>
#include <math.h>

// ---------------- problem sizes ----------------
#define BSZ     16384
#define IN_DIM  128
#define HID     512
#define ATT     256
#define TOTAL   896            // IN_DIM + HID + ATT
#define KBLKS   28             // TOTAL / 32
#define NT      32             // HID / 16  (n tiles)
#define BT      1024           // BSZ / 16  (b tiles)

// workspace layout (bytes)
#define CB_BYTES    ((size_t)BSZ * TOTAL * 2)          // combined bf16: 29,360,128
#define WGD_BYTES   ((size_t)(2 * HID) * TOTAL * 2)    // 1,835,008
#define WTAU_BYTES  ((size_t)HID * HID * 2)            // 524,288

typedef __attribute__((ext_vector_type(16))) __bf16 v16bf;
typedef __attribute__((ext_vector_type(8)))  __bf16 v8bf;
typedef __attribute__((ext_vector_type(8)))  float  v8f;

__device__ __forceinline__ float softplusf(float x) {
    // stable: max(x,0) + log1p(exp(-|x|))
    return fmaxf(x, 0.0f) + log1pf(expf(-fabsf(x)));
}
__device__ __forceinline__ float sigmoidf(float x) {
    return 1.0f / (1.0f + expf(-x));
}

// WMMA 16x16x32 bf16 A/B fragment: per lane, two contiguous 8-elem (16B) chunks
// at K offsets 8*(lane>>4) and 8*(lane>>4)+16 of row (lane&15).  [ISA 7.12.2]
__device__ __forceinline__ v16bf load_frag(const __bf16* __restrict__ p) {
    union { v16bf v; v8bf h[2]; } u;
    u.h[0] = *(const v8bf*)(p);
    u.h[1] = *(const v8bf*)(p + 16);
    return u.v;
}

// ---------------- prep kernels ----------------
__global__ void build_combined_k(const float* __restrict__ x,
                                 const float* __restrict__ h,
                                 const float* __restrict__ c,
                                 __bf16* __restrict__ cb) {
    const int b = blockIdx.x;
    const float* xr = x + (size_t)b * IN_DIM;
    const float* hr = h + (size_t)b * HID;
    const float* cr = c + (size_t)b * ATT;
    __bf16* o = cb + (size_t)b * TOTAL;
    for (int j = threadIdx.x; j < TOTAL; j += 256) {
        float v = (j < IN_DIM) ? xr[j]
                : (j < IN_DIM + HID) ? hr[j - IN_DIM]
                : cr[j - (IN_DIM + HID)];
        o[j] = (__bf16)v;
    }
}

__global__ void cvt_bf16_k(const float* __restrict__ src,
                           __bf16* __restrict__ dst, int n) {
    int i = blockIdx.x * 256 + threadIdx.x;
    if (i < n) dst[i] = (__bf16)src[i];
}

__global__ void make_spc_k(const float* __restrict__ gleak,
                           const float* __restrict__ cm,
                           float* __restrict__ spc) {
    int i = blockIdx.x * 256 + threadIdx.x;
    if (i < HID) spc[i] = softplusf(gleak[i]) + softplusf(cm[i]);
}

// ---------------- fused WMMA kernel ----------------
__global__ __launch_bounds__(256) void ltc_wmma_k(
        const __bf16* __restrict__ cb,    // [B, 896] bf16
        const __bf16* __restrict__ wgd,   // [1024, 896] bf16
        const __bf16* __restrict__ wtau,  // [512, 512] bf16
        const float*  __restrict__ spc,   // [512] softplus(cm)+softplus(gleak)
        const float*  __restrict__ h_ltc, // [B, 512] f32
        const float*  __restrict__ b_gd,  // [1024]
        const float*  __restrict__ b_tau, // [512]
        float* __restrict__ out)          // [B, 512]
{
    const int wave = threadIdx.x >> 5;
    const int lane = threadIdx.x & 31;
    const int tile = blockIdx.x * 8 + wave;      // 32768 tiles total
    const int n0 = (tile & (NT - 1)) * 16;       // 8 waves/block share b-tile
    const int b0 = (tile / NT) * 16;

    const int r   = lane & 15;                   // row-in-tile this lane loads
    const int hlf = lane >> 4;                   // K half selector

    const __bf16* aRow  = cb   + (size_t)(b0 + r)        * TOTAL + hlf * 8;
    const __bf16* b1Row = wgd  + (size_t)(n0 + r)        * TOTAL + hlf * 8;
    const __bf16* b2Row = wgd  + (size_t)(HID + n0 + r)  * TOTAL + hlf * 8;
    const __bf16* b3Row = wtau + (size_t)(n0 + r)        * HID   + hlf * 8;

    v8f accG = {}, accD = {}, accT = {};

    // K blocks 0..3: x_t region — gate + dynamics only
#pragma unroll
    for (int kb = 0; kb < 4; ++kb) {
        v16bf a  = load_frag(aRow  + kb * 32);
        v16bf w1 = load_frag(b1Row + kb * 32);
        v16bf w2 = load_frag(b2Row + kb * 32);
        accG = __builtin_amdgcn_wmma_f32_16x16x32_bf16(false, a, false, w1, (short)0, accG, false, false);
        accD = __builtin_amdgcn_wmma_f32_16x16x32_bf16(false, a, false, w2, (short)0, accD, false, false);
    }
    // K blocks 4..19: h_ltc region — gate + dynamics + tau share the A fragment
#pragma unroll 4
    for (int kb = 4; kb < 20; ++kb) {
        v16bf a  = load_frag(aRow  + kb * 32);
        v16bf w1 = load_frag(b1Row + kb * 32);
        v16bf w2 = load_frag(b2Row + kb * 32);
        v16bf w3 = load_frag(b3Row + (kb - 4) * 32);
        accG = __builtin_amdgcn_wmma_f32_16x16x32_bf16(false, a, false, w1, (short)0, accG, false, false);
        accD = __builtin_amdgcn_wmma_f32_16x16x32_bf16(false, a, false, w2, (short)0, accD, false, false);
        accT = __builtin_amdgcn_wmma_f32_16x16x32_bf16(false, a, false, w3, (short)0, accT, false, false);
    }
    // K blocks 20..27: context region — gate + dynamics only
#pragma unroll
    for (int kb = 20; kb < 28; ++kb) {
        v16bf a  = load_frag(aRow  + kb * 32);
        v16bf w1 = load_frag(b1Row + kb * 32);
        v16bf w2 = load_frag(b2Row + kb * 32);
        accG = __builtin_amdgcn_wmma_f32_16x16x32_bf16(false, a, false, w1, (short)0, accG, false, false);
        accD = __builtin_amdgcn_wmma_f32_16x16x32_bf16(false, a, false, w2, (short)0, accD, false, false);
    }

    // Epilogue. C/D layout: VGPR i, lane L -> M = i + 8*(L>>4), N = L&15.
    const int n = n0 + r;
    const float biasG  = b_gd[n];
    const float biasD  = b_gd[n + HID];
    const float biasT  = b_tau[n];
    const float dconst = spc[n] + 1e-6f;

    union { v8f v; float f[8]; } g, d, t;
    g.v = accG; d.v = accD; t.v = accT;

#pragma unroll
    for (int i = 0; i < 8; ++i) {
        const int b = b0 + i + 8 * hlf;
        float gate = g.f[i] + biasG;
        float dyn  = d.f[i] + biasD;
        float tau  = softplusf(t.f[i] + biasT);
        float hv   = h_ltc[(size_t)b * HID + n];
        float num  = sigmoidf(gate) * tanhf(dyn) - hv;
        out[(size_t)b * HID + n] = num / (tau + dconst);
    }
}

// ---------------- launch ----------------
extern "C" void kernel_launch(void* const* d_in, const int* in_sizes, int n_in,
                              void* d_out, int out_size, void* d_ws, size_t ws_size,
                              hipStream_t stream) {
    const float* h_ltc   = (const float*)d_in[1];
    const float* x_t     = (const float*)d_in[2];
    const float* context = (const float*)d_in[3];
    const float* W_gd    = (const float*)d_in[4];
    const float* b_gd    = (const float*)d_in[5];
    const float* W_tau   = (const float*)d_in[6];
    const float* b_tau   = (const float*)d_in[7];
    const float* gleak   = (const float*)d_in[8];
    const float* cm      = (const float*)d_in[9];
    float* out = (float*)d_out;

    char* ws = (char*)d_ws;
    __bf16* cb    = (__bf16*)(ws);
    __bf16* wgdb  = (__bf16*)(ws + CB_BYTES);
    __bf16* wtaub = (__bf16*)(ws + CB_BYTES + WGD_BYTES);
    float*  spc   = (float*)(ws + CB_BYTES + WGD_BYTES + WTAU_BYTES);

    // prep: bf16 conversions + per-channel denominator constant
    build_combined_k<<<BSZ, 256, 0, stream>>>(x_t, h_ltc, context, cb);
    {
        int n = 2 * HID * TOTAL;
        cvt_bf16_k<<<(n + 255) / 256, 256, 0, stream>>>(W_gd, wgdb, n);
    }
    {
        int n = HID * HID;
        cvt_bf16_k<<<(n + 255) / 256, 256, 0, stream>>>(W_tau, wtaub, n);
    }
    make_spc_k<<<2, 256, 0, stream>>>(gleak, cm, spc);

    // fused dual-GEMM + epilogue: 32768 tiles, 8 waves (tiles) per 256-thread block
    ltc_wmma_k<<<(BT * NT) / 8, 256, 0, stream>>>(cb, wgdb, wtaub, spc,
                                                  h_ltc, b_gd, b_tau, out);
}
